// HybridConditionEncoder_910533067527
// MI455X (gfx1250) — compile-verified, matrix operand order checked by reading, not run on hardware
//
#include <hip/hip_runtime.h>

// ---------------------------------------------------------------------------
// HybridConditionEncoder for MI455X (gfx1250, wave32, WMMA).
//
// Sizes (fixed by reference config):
//   B=16384, N_DS=8 (4 single + 4 multi), N_CLS_S=4096, N_CLS_M=1024,
//   D_DS=128, D_CLS=256, D_OUT=256.
//
// Memory-bound problem (~90 MB of traffic -> ~4us at 23.3 TB/s). Strategy:
//   * bucket-compact multi-label samples per encoder, GEMM only the selected
//     encoder (reference computes all 4 redundantly),
//   * f16 activations/weights + f32 accumulation on v_wmma_f32_16x16x32_f16,
//   * cond intermediate kept in f16 (halves traffic vs f32).
// ---------------------------------------------------------------------------

typedef __attribute__((ext_vector_type(16))) _Float16 v16h;
typedef __attribute__((ext_vector_type(8)))  _Float16 v8h;
typedef __attribute__((ext_vector_type(8)))  float    v8f;

union V16 { v16h v; v8h h[2]; };

#define NB        16384
#define D_DS      128
#define D_CLS     256
#define D_IN      384      // D_DS + D_CLS
#define D_OUT     256
#define N_CLS_S   4096
#define N_CLS_M   1024
#define N_SINGLE  4
#define N_MULTI   4

__device__ __forceinline__ v8f wmma_f16(v16h a, v16h b, v8f c) {
  // (neg_a, A, neg_b, B, c_mod, C, reuse_a, reuse_b)
  return __builtin_amdgcn_wmma_f32_16x16x32_f16(false, a, false, b,
                                                (short)0, c, false, false);
}

// ---------------------------------------------------------------------------
// Kernel 0: convert W_multi [4,256,1024] and W_proj [256,384] to f16 copies.
// ---------------------------------------------------------------------------
__global__ void convert_weights_kernel(const float* __restrict__ Wm,
                                       const float* __restrict__ Wp,
                                       _Float16* __restrict__ Wm16,
                                       _Float16* __restrict__ Wp16) {
  const int nWm = N_MULTI * D_CLS * N_CLS_M;   // 1,048,576
  const int nWp = D_OUT * D_IN;                // 98,304
  int idx = blockIdx.x * blockDim.x + threadIdx.x;
  int stride = gridDim.x * blockDim.x;
  for (int i = idx; i < nWm; i += stride) Wm16[i] = (_Float16)Wm[i];
  for (int i = idx; i < nWp; i += stride) Wp16[i] = (_Float16)Wp[i];
}

// ---------------------------------------------------------------------------
// Kernel 1: per-sample prep. One block (128 threads) per sample.
//   cond[b][0:128]   = E_ds[dataset_id]
//   single samples:  cond[b][128:384] = E_single[ds][label]
//   multi samples:   cond[b][128:384] = 0 (filled by GEMM), bucket-compact b.
// ---------------------------------------------------------------------------
__global__ void prep_kernel(const int* __restrict__ dataset_ids,
                            const int* __restrict__ single_labels,
                            const float* __restrict__ E_ds,
                            const float* __restrict__ E_single,
                            _Float16* __restrict__ cond,
                            int* __restrict__ counts,
                            int* __restrict__ buckets) {
  int b = blockIdx.x;
  int t = threadIdx.x;               // 0..127
  int did = dataset_ids[b];
  _Float16* crow = cond + (size_t)b * D_IN;
  crow[t] = (_Float16)E_ds[did * D_DS + t];
  if (did < N_SINGLE) {
    int lab = single_labels[b];
    const float* src = E_single + ((size_t)did * N_CLS_S + lab) * D_CLS;
    crow[D_DS + t]       = (_Float16)src[t];
    crow[D_DS + 128 + t] = (_Float16)src[t + 128];
  } else {
    crow[D_DS + t]       = (_Float16)0.f;
    crow[D_DS + 128 + t] = (_Float16)0.f;
    if (t == 0) {
      int e = did - N_SINGLE;
      int pos = atomicAdd(&counts[e], 1);
      buckets[e * NB + pos] = b;
    }
  }
}

// ---------------------------------------------------------------------------
// Kernel 2: per-bucket gathered GEMM.
//   multi_vec[tile 16 rows][256] = A(16x1024) x B(1024x256) + bias,
//   B[k][n] = W_multi[e][n][k] (f16 copy), scatter into cond[:,128:384].
// Block: 256 threads (8 waves); wave w owns N-tiles 2w, 2w+1.
// ---------------------------------------------------------------------------
__global__ void multi_gemm_kernel(const float* __restrict__ multi_labels,
                                  const _Float16* __restrict__ Wm16,
                                  const float* __restrict__ b_multi,
                                  const int* __restrict__ counts,
                                  const int* __restrict__ buckets,
                                  _Float16* __restrict__ cond) {
  __shared__ _Float16 Als[16][N_CLS_M];   // 32 KB f16 A tile
  __shared__ int rows[16];

  const int e = blockIdx.y;
  const int cnt = counts[e];
  const int m0 = blockIdx.x * 16;
  if (m0 >= cnt) return;

  const int tid = threadIdx.x;
  if (tid < 16) {
    int m = m0 + tid;
    rows[tid] = (m < cnt) ? buckets[e * NB + m] : -1;
  }
  __syncthreads();

  // Stage A: gather 16 rows x 1024 f32 -> f16 LDS. 4096 float4 / 256 thr.
  for (int i = tid; i < 16 * (N_CLS_M / 4); i += 256) {
    int r = i >> 8;               // row 0..15
    int c = (i & 255) << 2;       // col, step 4
    float4 v = make_float4(0.f, 0.f, 0.f, 0.f);
    int gb = rows[r];
    if (gb >= 0) v = *(const float4*)(multi_labels + (size_t)gb * N_CLS_M + c);
    Als[r][c + 0] = (_Float16)v.x;
    Als[r][c + 1] = (_Float16)v.y;
    Als[r][c + 2] = (_Float16)v.z;
    Als[r][c + 3] = (_Float16)v.w;
  }
  __syncthreads();

  const int lane = tid & 31;
  const int wave = tid >> 5;
  const int l16  = lane & 15;
  const int koff = (lane >> 4) << 3;     // 0 or 8
  const int nb0 = wave * 32;
  const int nb1 = nb0 + 16;

  const _Float16* Ar    = &Als[l16][0];
  const _Float16* Brow0 = Wm16 + ((size_t)(e * D_CLS + nb0 + l16)) * N_CLS_M;
  const _Float16* Brow1 = Wm16 + ((size_t)(e * D_CLS + nb1 + l16)) * N_CLS_M;

  v8f acc0 = {}; v8f acc1 = {};
  for (int k0 = 0; k0 < N_CLS_M; k0 += 32) {
    __builtin_prefetch(Brow0 + k0 + 256, 0, 1);
    __builtin_prefetch(Brow1 + k0 + 256, 0, 1);
    V16 a, b0, b1;
    a.h[0]  = *(const v8h*)(Ar + k0 + koff);
    a.h[1]  = *(const v8h*)(Ar + k0 + 16 + koff);
    b0.h[0] = *(const v8h*)(Brow0 + k0 + koff);
    b0.h[1] = *(const v8h*)(Brow0 + k0 + 16 + koff);
    b1.h[0] = *(const v8h*)(Brow1 + k0 + koff);
    b1.h[1] = *(const v8h*)(Brow1 + k0 + 16 + koff);
    acc0 = wmma_f16(a.v, b0.v, acc0);
    acc1 = wmma_f16(a.v, b1.v, acc1);
  }

  const int n0 = nb0 + l16;
  const int n1 = nb1 + l16;
  const float bias0 = b_multi[e * D_CLS + n0];
  const float bias1 = b_multi[e * D_CLS + n1];
  #pragma unroll
  for (int r = 0; r < 8; ++r) {
    int m = r + ((lane >> 4) << 3);
    int gb = rows[m];
    if (gb >= 0) {
      _Float16* crow = cond + (size_t)gb * D_IN + D_DS;
      crow[n0] = (_Float16)(acc0[r] + bias0);
      crow[n1] = (_Float16)(acc1[r] + bias1);
    }
  }
}

// ---------------------------------------------------------------------------
// Kernel 3: projection GEMM. out[B,256] = cond[B,384] x W_proj^T + b_proj.
//   B[k][n] = W_proj[n][k] (f16 copy). One block per 16-sample M tile;
//   8 waves x 2 N-tiles cover N=256.
// ---------------------------------------------------------------------------
__global__ void proj_gemm_kernel(const _Float16* __restrict__ cond,
                                 const _Float16* __restrict__ Wp16,
                                 const float* __restrict__ b_proj,
                                 float* __restrict__ out) {
  const int m0   = blockIdx.x * 16;
  const int tid  = threadIdx.x;
  const int lane = tid & 31;
  const int wave = tid >> 5;
  const int l16  = lane & 15;
  const int koff = (lane >> 4) << 3;
  const int nb0 = wave * 32;
  const int nb1 = nb0 + 16;

  const _Float16* Arow  = cond + (size_t)(m0 + l16) * D_IN;
  const _Float16* Brow0 = Wp16 + (size_t)(nb0 + l16) * D_IN;
  const _Float16* Brow1 = Wp16 + (size_t)(nb1 + l16) * D_IN;

  v8f acc0 = {}; v8f acc1 = {};
  #pragma unroll
  for (int k0 = 0; k0 < D_IN; k0 += 32) {
    V16 a, b0, b1;
    a.h[0]  = *(const v8h*)(Arow + k0 + koff);
    a.h[1]  = *(const v8h*)(Arow + k0 + 16 + koff);
    b0.h[0] = *(const v8h*)(Brow0 + k0 + koff);
    b0.h[1] = *(const v8h*)(Brow0 + k0 + 16 + koff);
    b1.h[0] = *(const v8h*)(Brow1 + k0 + koff);
    b1.h[1] = *(const v8h*)(Brow1 + k0 + 16 + koff);
    acc0 = wmma_f16(a.v, b0.v, acc0);
    acc1 = wmma_f16(a.v, b1.v, acc1);
  }

  const int n0 = nb0 + l16;
  const int n1 = nb1 + l16;
  const float bias0 = b_proj[n0];
  const float bias1 = b_proj[n1];
  #pragma unroll
  for (int r = 0; r < 8; ++r) {
    int m = m0 + r + ((lane >> 4) << 3);
    out[(size_t)m * D_OUT + n0] = acc0[r] + bias0;
    out[(size_t)m * D_OUT + n1] = acc1[r] + bias1;
  }
}

// ---------------------------------------------------------------------------
// Launch. Workspace layout (all 256B aligned):
//   [0,16)            counts (4 int)
//   [256, +256KB)     buckets (4 x 16384 int)
//   [+2MB)            Wm16  (4*256*1024 f16)
//   [+192KB)          Wp16  (256*384 f16)
//   [+12MB)           cond  (16384*384 f16)
// Total ~14.5 MB.
// ---------------------------------------------------------------------------
extern "C" void kernel_launch(void* const* d_in, const int* in_sizes, int n_in,
                              void* d_out, int out_size, void* d_ws, size_t ws_size,
                              hipStream_t stream) {
  const int*   dataset_ids   = (const int*)d_in[0];
  const int*   single_labels = (const int*)d_in[1];
  const float* multi_labels  = (const float*)d_in[2];
  const float* E_ds          = (const float*)d_in[3];
  const float* E_single      = (const float*)d_in[4];
  const float* W_multi       = (const float*)d_in[5];
  const float* b_multi       = (const float*)d_in[6];
  const float* W_proj        = (const float*)d_in[7];
  const float* b_proj        = (const float*)d_in[8];
  float* out = (float*)d_out;

  char* ws = (char*)d_ws;
  int*      counts  = (int*)ws;
  int*      buckets = (int*)(ws + 256);
  _Float16* Wm16    = (_Float16*)(ws + 256 + (size_t)N_MULTI * NB * sizeof(int));
  _Float16* Wp16    = Wm16 + (size_t)N_MULTI * D_CLS * N_CLS_M;
  _Float16* cond    = Wp16 + (size_t)D_OUT * D_IN;

  hipMemsetAsync(counts, 0, N_MULTI * sizeof(int), stream);

  convert_weights_kernel<<<1024, 256, 0, stream>>>(W_multi, W_proj, Wm16, Wp16);

  prep_kernel<<<NB, 128, 0, stream>>>(dataset_ids, single_labels, E_ds, E_single,
                                      cond, counts, buckets);

  // Worst case one bucket holds all B samples -> NB/16 tiles per bucket.
  multi_gemm_kernel<<<dim3(NB / 16, N_MULTI), 256, 0, stream>>>(
      multi_labels, Wm16, b_multi, counts, buckets, cond);

  proj_gemm_kernel<<<NB / 16, 256, 0, stream>>>(cond, Wp16, b_proj, out);
}
